// AttentiveAggregator_75488345195336
// MI455X (gfx1250) — matrix operand order repaired and checked by reading, not compiled, not run on hardware
//
#include <hip/hip_runtime.h>
#include <hip/hip_bf16.h>
#include <math.h>

typedef _Float16 v16h __attribute__((ext_vector_type(16)));
typedef float    v8f  __attribute__((ext_vector_type(8)));

#define NWAVES 8   // waves (nodes) per 256-thread block

union F16x16 { v16h h; uint4 q[2]; };
union F16x4  { _Float16 h[4]; uint2 u; };

// Branchless exact-erf GELU: 0.5x(1+erf(x/sqrt2)); A&S 7.1.26, |err|<1.5e-7.
// ~13 VALU, no EXEC divergence (keeps scheduling around WMMA clean).
__device__ __forceinline__ float gelu_exact(float x) {
  float z = x * 0.70710678118654752f;
  float a = __builtin_fabsf(z);
  float t = __builtin_amdgcn_rcpf(__builtin_fmaf(0.3275911f, a, 1.0f));
  float p = t * __builtin_fmaf(t, __builtin_fmaf(t, __builtin_fmaf(t,
                __builtin_fmaf(t, 1.061405429f, -1.453152027f),
                1.421413741f), -0.284496736f), 0.254829592f);
  float e = __expf(-a * a);
  float erfa = __builtin_fmaf(-p, e, 1.0f);          // erf(|z|)
  float erfz = __builtin_copysignf(erfa, z);
  return 0.5f * x * (1.0f + erfz);
}

__device__ __forceinline__ float sigmoid_fast(float x) {
  return __builtin_amdgcn_rcpf(1.0f + __expf(-x));
}

// ---------------------------------------------------------------------------
// K0: zero per-node counts + repack W1 (64x256, torch [out,in]) into WMMA
//     B-fragment order: frag(c,t,lane)[e] = W1[16t+(lane&15)][32c+(lane&16)+e]
// ---------------------------------------------------------------------------
__global__ void aa_init_kernel(unsigned* __restrict__ counts,
                               _Float16* __restrict__ w1frag,
                               const float* __restrict__ W1, int n_nodes) {
  int j = blockIdx.x * blockDim.x + threadIdx.x;
  if (j < n_nodes) counts[j] = 0u;
  if (j < 1024) {               // 8 k-chunks * 4 n-tiles * 32 lanes
    int l = j & 31;
    int t = (j >> 5) & 3;
    int c = j >> 7;
    int ncol = t * 16 + (l & 15);
    int kb   = c * 32 + (l & 16);
#pragma unroll
    for (int e = 0; e < 16; ++e)
      w1frag[j * 16 + e] = (_Float16)W1[ncol * 256 + kb + e];
  }
}

// K1: histogram of target indices
__global__ void aa_hist_kernel(const int* __restrict__ tgt,
                               unsigned* __restrict__ counts, int E) {
  int e = blockIdx.x * blockDim.x + threadIdx.x;
  if (e < E) atomicAdd(&counts[tgt[e]], 1u);
}

// K2: single-block exclusive scan (chunked Hillis-Steele), 1024 threads
__global__ void aa_scan_kernel(const unsigned* __restrict__ counts,
                               unsigned* __restrict__ offsets,
                               unsigned* __restrict__ cursor, int n) {
  __shared__ unsigned buf[1024];
  int tid = threadIdx.x;
  unsigned carry = 0;
  for (int base = 0; base < n; base += 1024) {
    int i = base + tid;
    unsigned v = (i < n) ? counts[i] : 0u;
    buf[tid] = v;
    __syncthreads();
    for (int off = 1; off < 1024; off <<= 1) {
      unsigned t = (tid >= off) ? buf[tid - off] : 0u;
      __syncthreads();
      buf[tid] += t;
      __syncthreads();
    }
    unsigned incl = buf[tid];
    unsigned tot  = buf[1023];
    if (i < n) { unsigned ex = carry + incl - v; offsets[i] = ex; cursor[i] = ex; }
    carry += tot;
    __syncthreads();
  }
  if (tid == 0) offsets[n] = carry;
}

// K3: CSR fill (edge id -> its node's slot)
__global__ void aa_fill_kernel(const int* __restrict__ tgt,
                               unsigned* __restrict__ cursor,
                               unsigned* __restrict__ edge_ids, int E) {
  int e = blockIdx.x * blockDim.x + threadIdx.x;
  if (e < E) {
    unsigned pos = atomicAdd(&cursor[tgt[e]], 1u);
    edge_ids[pos] = (unsigned)e;
  }
}

// ---------------------------------------------------------------------------
// K4: fused per-node kernel. One wave per node. c-outer / t-inner WMMA order
// keeps only one A-fragment live at a time; B-fragments are reloaded from the
// (L0/L2-resident) repacked W1 each chunk via an opaque *integer offset* so
// the compiler can neither hoist 256 VGPRs of weights out of the tile loop
// nor lose the global address space (global_load_b128, not flat_load_b128 —
// flat would tie up the LDS pipe needed by the A-fragment ds_loads).
// ---------------------------------------------------------------------------
__global__ __launch_bounds__(256) void aa_node_kernel(
    const float* __restrict__ messages,
    const float* __restrict__ node_features,
    const float* __restrict__ b1,
    const float* __restrict__ W2,
    const float* __restrict__ gamma,
    const float* __restrict__ beta,
    const unsigned* __restrict__ offsets,
    const unsigned* __restrict__ edge_ids,
    const _Float16* __restrict__ w1frag,
    float* __restrict__ out, int n_nodes) {
  __shared__ _Float16 sMsg[NWAVES][16][128];
  __shared__ _Float16 sFeat[NWAVES][128];

  const int lane = threadIdx.x & 31;
  const int wave = threadIdx.x >> 5;
  const int node = blockIdx.x * NWAVES + wave;
  if (node >= n_nodes) return;          // wave-uniform: WMMA below has full EXEC

  // stage this node's features once (K = 128..255 half of the concat)
  {
    float4 f = *(const float4*)&node_features[(size_t)node * 128 + lane * 4];
    F16x4 p; p.h[0] = (_Float16)f.x; p.h[1] = (_Float16)f.y;
             p.h[2] = (_Float16)f.z; p.h[3] = (_Float16)f.w;
    *(uint2*)&sFeat[wave][lane * 4] = p.u;
  }
  __builtin_amdgcn_wave_barrier();

  const int kbA  = ((lane >> 4) << 4);  // +16 for upper 16 lanes (contiguous k, matches w1frag)
  const int ncol = lane & 15;           // D-matrix N column for this lane
  const int row  = lane & 15;
  float b1v[4], w2v[4];
#pragma unroll
  for (int t = 0; t < 4; ++t) { b1v[t] = b1[t * 16 + ncol]; w2v[t] = W2[t * 16 + ncol]; }

  const unsigned beg = offsets[node];
  const int deg = (int)(offsets[node + 1] - beg);

  float4 acc = make_float4(0.f, 0.f, 0.f, 0.f);
  float wsum = 0.f;

  for (int ts = 0; ts < deg; ts += 16) {
    // ---- stage up to 16 messages as f16: batch ids, batch loads, then store
    unsigned eids[16];
#pragma unroll
    for (int e = 0; e < 16; ++e)
      eids[e] = (ts + e < deg) ? edge_ids[beg + ts + e] : 0u;  // uniform addr -> SMEM
#pragma unroll
    for (int e = 0; e < 16; ++e) {
      float4 m = make_float4(0.f, 0.f, 0.f, 0.f);
      if (ts + e < deg)                                        // wave-uniform branch
        m = *(const float4*)&messages[(size_t)eids[e] * 128 + lane * 4];
      F16x4 p; p.h[0] = (_Float16)m.x; p.h[1] = (_Float16)m.y;
               p.h[2] = (_Float16)m.z; p.h[3] = (_Float16)m.w;
      *(uint2*)&sMsg[wave][e][lane * 4] = p.u;
    }
    __builtin_amdgcn_wave_barrier();

    // ---- WMMA: c-outer (one A-frag live), t-inner (4 accumulators) ----
    v8f d[4];
#pragma unroll
    for (int t = 0; t < 4; ++t) d[t] = (v8f){0.f,0.f,0.f,0.f,0.f,0.f,0.f,0.f};
#pragma unroll
    for (int c = 0; c < 8; ++c) {
      F16x16 a;
      if (c < 4) {
        a.q[0] = *(const uint4*)&sMsg[wave][row][c * 32 + kbA];
        a.q[1] = *(const uint4*)&sMsg[wave][row][c * 32 + kbA + 8];
      } else {
        a.q[0] = *(const uint4*)&sFeat[wave][(c - 4) * 32 + kbA];
        a.q[1] = *(const uint4*)&sFeat[wave][(c - 4) * 32 + kbA + 8];
      }
      // opaque integer offset: defeats hoisting/CSE of the B-fragment loads
      // while keeping w1frag in the global address space.
      unsigned fbias = 0;
      asm volatile("" : "+s"(fbias));
      const uint4* bp = (const uint4*)w1frag + fbias;
#pragma unroll
      for (int t = 0; t < 4; ++t) {
        int fi = ((c * 4 + t) * 32 + lane) * 2;
        F16x16 b; b.q[0] = bp[fi]; b.q[1] = bp[fi + 1];
        d[t] = __builtin_amdgcn_wmma_f32_16x16x32_f16(
            false, a.h, false, b.h, (short)0, d[t], false, false);
      }
    }

    // ---- score: sum_N gelu(h+b1)*W2[N]; rows 0..7 in lanes 0..15, 8..15 above
    float s[8];
#pragma unroll
    for (int v = 0; v < 8; ++v) s[v] = 0.f;
#pragma unroll
    for (int t = 0; t < 4; ++t)
#pragma unroll
      for (int v = 0; v < 8; ++v)
        s[v] = __builtin_fmaf(gelu_exact(d[t][v] + b1v[t]), w2v[t], s[v]);
#pragma unroll
    for (int msk = 1; msk < 16; msk <<= 1)
#pragma unroll
      for (int v = 0; v < 8; ++v) s[v] += __shfl_xor(s[v], msk, 32);
    float sg[8];
#pragma unroll
    for (int v = 0; v < 8; ++v) sg[v] = sigmoid_fast(s[v]);

    // ---- accumulate weighted f32 messages (re-read hits L0; ids reused) ----
    float4 mm[16];
#pragma unroll
    for (int e = 0; e < 16; ++e) {
      mm[e] = make_float4(0.f, 0.f, 0.f, 0.f);
      if (ts + e < deg)
        mm[e] = *(const float4*)&messages[(size_t)eids[e] * 128 + lane * 4];
    }
#pragma unroll
    for (int e = 0; e < 16; ++e) {
      float we = __shfl(sg[e & 7], (e & 8) ? 16 : 0, 32);
      if (ts + e >= deg) we = 0.f;
      acc.x = __builtin_fmaf(we, mm[e].x, acc.x);
      acc.y = __builtin_fmaf(we, mm[e].y, acc.y);
      acc.z = __builtin_fmaf(we, mm[e].z, acc.z);
      acc.w = __builtin_fmaf(we, mm[e].w, acc.w);
      wsum += we;
    }
    __builtin_amdgcn_wave_barrier();
  }

  // ---- mean-normalize + LayerNorm over 128 (32-lane shuffle reduction) ----
  float inv = 1.0f / (wsum + 1e-8f);
  float x0 = acc.x * inv, x1 = acc.y * inv, x2 = acc.z * inv, x3 = acc.w * inv;
  float ls = x0 + x1 + x2 + x3;
  float lq = x0 * x0 + x1 * x1 + x2 * x2 + x3 * x3;
#pragma unroll
  for (int msk = 1; msk < 32; msk <<= 1) {
    ls += __shfl_xor(ls, msk, 32);
    lq += __shfl_xor(lq, msk, 32);
  }
  float mean = ls * (1.0f / 128.0f);
  float var  = lq * (1.0f / 128.0f) - mean * mean;
  float rstd = rsqrtf(var + 1e-5f);
  float4 g4 = *(const float4*)&gamma[lane * 4];
  float4 b4 = *(const float4*)&beta[lane * 4];
  float4 o;
  o.x = (x0 - mean) * rstd * g4.x + b4.x;
  o.y = (x1 - mean) * rstd * g4.y + b4.y;
  o.z = (x2 - mean) * rstd * g4.z + b4.z;
  o.w = (x3 - mean) * rstd * g4.w + b4.w;
  *(float4*)&out[(size_t)node * 128 + lane * 4] = o;
}

// ---------------------------------------------------------------------------
extern "C" void kernel_launch(void* const* d_in, const int* in_sizes, int n_in,
                              void* d_out, int out_size, void* d_ws, size_t ws_size,
                              hipStream_t stream) {
  const float* messages = (const float*)d_in[0];
  const int*   tgt      = (const int*)d_in[1];
  const float* nodef    = (const float*)d_in[2];
  // d_in[3] = n_nodes scalar (unused; derived from sizes)
  const float* W1       = (const float*)d_in[4];
  const float* b1       = (const float*)d_in[5];
  const float* W2       = (const float*)d_in[6];
  const float* gamma    = (const float*)d_in[7];
  const float* beta     = (const float*)d_in[8];
  float* out = (float*)d_out;

  const int E = in_sizes[0] / 128;
  const int N = in_sizes[2] / 128;

  char* ws = (char*)d_ws;
  size_t off = 0;
  auto alloc = [&](size_t bytes) { char* p = ws + off; off += (bytes + 255) & ~(size_t)255; return p; };
  unsigned* counts   = (unsigned*)alloc((size_t)N * 4);
  unsigned* offsets  = (unsigned*)alloc((size_t)(N + 1) * 4);
  unsigned* cursor   = (unsigned*)alloc((size_t)N * 4);
  unsigned* edge_ids = (unsigned*)alloc((size_t)E * 4);
  _Float16* w1frag   = (_Float16*)alloc(16384 * 2);
  (void)ws_size;

  int initN = (N > 1024) ? N : 1024;
  aa_init_kernel<<<(initN + 255) / 256, 256, 0, stream>>>(counts, w1frag, W1, N);
  aa_hist_kernel<<<(E + 255) / 256, 256, 0, stream>>>(tgt, counts, E);
  aa_scan_kernel<<<1, 1024, 0, stream>>>(counts, offsets, cursor, N);
  aa_fill_kernel<<<(E + 255) / 256, 256, 0, stream>>>(tgt, cursor, edge_ids, E);
  aa_node_kernel<<<(N + NWAVES - 1) / NWAVES, 256, 0, stream>>>(
      messages, nodef, b1, W2, gamma, beta, offsets, edge_ids, w1frag, out, N);
}
